// LTIRouter_19679540150543
// MI455X (gfx1250) — compile-verified, matrix operand order checked by reading, not run on hardware
//
#include <hip/hip_runtime.h>
#include <hip/hip_bf16.h>

// ---------------------------------------------------------------------------
// Block-sparse causal IRF routing on MI455X (gfx1250).
//
// y[b, r*16+i, t] = x[b, r*16+i, t]
//                 + sum_blocks(n: rows==r) sum_j sum_k h(n,i,j,k) * x[b, c*16+j, t-k]
// h(n,i,j,k) = (1-a)*a^k, a = exp(-1/(1+19*params)), k < 100.
//
// v_wmma_f32_16x16x32_bf16 formulation: inner dim p = k*16 + j (k-major,
// taps padded to 104 -> 1664), A = IRF matrix H[16 x 1664] in LDS, B =
// time-reversed sliding window of x in LDS, f32 accumulators.
//
// Per wave: two 16-col time tiles, 16 steps apart. Since each wmma k-step
// slides the window by 2 taps, tile1's B fragment at step kk equals tile0's
// fragment at step kk-8 -> B fragments are pipelined through an 8-deep VGPR
// ring (2 ds_load_b128 per wmma total). A and the fresh B are additionally
// software-pipelined one k-step ahead so every wmma consumes operands that
// are already in registers (waits become dscnt<=4 instead of dscnt==0).
// ---------------------------------------------------------------------------

typedef __attribute__((ext_vector_type(16))) __bf16 v16bf;
typedef __attribute__((ext_vector_type(8)))  __bf16 v8bf;
typedef __attribute__((ext_vector_type(8)))  float  v8f;

#define BB     4
#define CC     4096
#define TT     2048
#define BSZ    16
#define KTAP   100
#define KP     104                 // taps padded to multiple of 8
#define KIN    (KP * 16)           // 1664 = padded inner (j,k) dimension
#define NNZB   1024
#define CT     256                 // timesteps per chunk (8 waves x 2 x 16)
#define HIST   KP                  // history needed per chunk
#define NCHUNK (TT / CT)           // 8
#define KSTEPS (KIN / 32)          // 52 wmma k-steps

__device__ __forceinline__ __bf16 f2bf(float f) {
  unsigned u = __builtin_bit_cast(unsigned, f);
  unsigned r = u + 0x7FFFu + ((u >> 16) & 1u);   // round-to-nearest-even
  unsigned short h = (unsigned short)(r >> 16);
  return __builtin_bit_cast(__bf16, h);
}

// B fragment: rows (taps) row,row+1 of the reversed window, j = sel..sel+7
// per half-wave; two aligned 16-byte LDS runs.
__device__ __forceinline__ v16bf load_bfrag(const __bf16 (*__restrict__ Xst)[16],
                                            int row, int sel) {
  const v8bf lo = *(const v8bf*)&Xst[row][sel];
  const v8bf hi = *(const v8bf*)&Xst[row + 1][sel];
  return __builtin_shufflevector(lo, hi, 0, 1, 2, 3, 4, 5, 6, 7,
                                 8, 9, 10, 11, 12, 13, 14, 15);
}

// A fragment: row = nn, inner runs kk*32+sel..+7 and +16..+23 (b128 each).
__device__ __forceinline__ v16bf load_afrag(const __bf16 (*__restrict__ Hs)[KIN],
                                            int kk, int row, int sel) {
  const v8bf lo = *(const v8bf*)&Hs[row][kk * 32 + sel];
  const v8bf hi = *(const v8bf*)&Hs[row][kk * 32 + sel + 16];
  return __builtin_shufflevector(lo, hi, 0, 1, 2, 3, 4, 5, 6, 7,
                                 8, 9, 10, 11, 12, 13, 14, 15);
}

__global__ void __launch_bounds__(256)
lti_route_kernel(const float* __restrict__ runoff,
                 const float* __restrict__ params,
                 const int*   __restrict__ brow,
                 const int*   __restrict__ bcol,
                 float*       __restrict__ out) {
  // 52 KB IRF matrix + 11.25 KB reversed-time x window = 63.25 KB LDS
  __shared__ __align__(16) __bf16 Hs[16][KIN];
  __shared__ __align__(16) __bf16 Xst[CT + HIST][16];

  const int n    = blockIdx.x;       // nonzero block id
  const int b    = blockIdx.y;       // batch
  const int tid  = threadIdx.x;
  const int lane = tid & 31;
  const int wv   = tid >> 5;         // wave id 0..7 -> 32-col slab in chunk

  const int r = brow[n];
  const int c = bcol[n];

  // ---- Build H[i][k*16+j] = (1-a) * a^k (bf16), zero-padded taps -------
  {
    const int i = tid >> 4, j = tid & 15;
    const float p   = params[n * 256 + i * 16 + j];
    const float tau = 1.0f + 19.0f * p;
    const float a   = __expf(-1.0f / tau);
    float s = 1.0f - a;
#pragma unroll 4
    for (int k = 0; k < KTAP; ++k) { Hs[i][k * 16 + j] = f2bf(s); s *= a; }
    for (int k = KTAP; k < KP; ++k) { Hs[i][k * 16 + j] = f2bf(0.0f); }
  }

  const float* xsrc = runoff + ((size_t)b * CC + (size_t)c * BSZ) * TT;
  float*       ydst = out    + ((size_t)b * CC + (size_t)r * BSZ) * TT;

  // WMMA fragment geometry (16-bit A/B layout, wave32):
  // lane<16 holds K-offsets {0..7, 16..23}; lane>=16 holds {8..15, 24..31}.
  const int sel   = (lane < 16) ? 0 : 8;
  const int nn    = lane & 15;               // B column / A row within tile
  const int mbase = (lane >= 16) ? 8 : 0;    // C/D row offset

  for (int ch = 0; ch < NCHUNK; ++ch) {
    const int t0 = ch * CT;
    __syncthreads();   // H ready / previous chunk's B reads done

    // ---- Load time-reversed window: Xst[u][j] = x_j[t0 + CT-1 - u] -----
    for (int e = tid; e < 16 * (CT + HIST); e += 256) {
      const int j = e / (CT + HIST);
      const int u = e - j * (CT + HIST);
      const int t = t0 + CT - 1 - u;
      const float v = (t >= 0) ? xsrc[(size_t)j * TT + t] : 0.0f;
      Xst[u][j] = f2bf(v);
      // speculative prefetch of next chunk (global_prefetch_b8)
      if (((e & 63) == 0) && (t >= 0) && (t + CT < TT))
        __builtin_prefetch(&xsrc[(size_t)j * TT + t + CT], 0, 1);
    }
    __syncthreads();

    // ---- 16x1664 (A) x 1664x32 (two B tiles), f32 accumulate -----------
    // tile0 cols: t0 + wv*32 + nn ; tile1 cols: +16.
    // Identity: Xst[u1 + 2kk] == Xst[u0 + 2(kk-8)] -> reuse B fragments
    // across tiles via an 8-deep register ring.
    const int u0 = CT - 1 - wv * 32 - nn;    // time-reverse offset, tile0
    const int u1 = u0 - 16;                  // tile1
    v8f acc0 = {};
    v8f acc1 = {};

    v16bf bbuf[8];
#pragma unroll
    for (int m = 0; m < 8; ++m)              // prologue: tile1 steps 0..7
      bbuf[m] = load_bfrag(Xst, u1 + 2 * m, sel);

    // 1-step software pipeline: A and fresh-B for step 0 preloaded here.
    v16bf av_cur = load_afrag(Hs, 0, nn, sel);
    v16bf bn_cur = load_bfrag(Xst, u0, sel);

#pragma unroll
    for (int kk = 0; kk < KSTEPS; ++kk) {
      v16bf av_nxt = av_cur;
      v16bf bn_nxt = bn_cur;
      if (kk + 1 < KSTEPS) {                 // issue next step's 4 loads now
        av_nxt = load_afrag(Hs, kk + 1, nn, sel);
        bn_nxt = load_bfrag(Xst, u0 + 2 * (kk + 1), sel);
      }
      // Both wmmas consume register-resident operands (loaded last step).
      acc1 = __builtin_amdgcn_wmma_f32_16x16x32_bf16(
          false, av_cur, false, bbuf[kk & 7], (short)0, acc1, false, false);
      acc0 = __builtin_amdgcn_wmma_f32_16x16x32_bf16(
          false, av_cur, false, bn_cur, (short)0, acc0, false, false);
      bbuf[kk & 7] = bn_cur;                 // becomes tile1's step kk+8
      av_cur = av_nxt;
      bn_cur = bn_nxt;
    }

    // ---- Scatter-add into output row block (rows collide across n) -----
    const int tcol0 = t0 + wv * 32 + nn;
#pragma unroll
    for (int v = 0; v < 8; ++v)
      atomicAdd(&ydst[(size_t)(v + mbase) * TT + tcol0], acc0[v]);
#pragma unroll
    for (int v = 0; v < 8; ++v)
      atomicAdd(&ydst[(size_t)(v + mbase) * TT + tcol0 + 16], acc1[v]);
  }
}

// out = runoff (identity term); also re-initializes out every call so the
// routing kernel's atomics accumulate onto a fresh base (determinism).
__global__ void __launch_bounds__(256)
lti_init_kernel(const float4* __restrict__ src, float4* __restrict__ dst, int n4) {
  const int i = blockIdx.x * blockDim.x + threadIdx.x;
  if (i < n4) dst[i] = src[i];
}

extern "C" void kernel_launch(void* const* d_in, const int* in_sizes, int n_in,
                              void* d_out, int out_size, void* d_ws, size_t ws_size,
                              hipStream_t stream) {
  (void)in_sizes; (void)n_in; (void)out_size; (void)d_ws; (void)ws_size;
  const float* runoff = (const float*)d_in[0];
  const float* params = (const float*)d_in[1];
  const int*   brow   = (const int*)d_in[2];
  const int*   bcol   = (const int*)d_in[3];
  float*       out    = (float*)d_out;

  const int n4 = (BB * CC * TT) / 4;
  lti_init_kernel<<<(n4 + 255) / 256, 256, 0, stream>>>(
      (const float4*)runoff, (float4*)out, n4);

  dim3 grid(NNZB, BB);
  lti_route_kernel<<<grid, 256, 0, stream>>>(runoff, params, brow, bcol, out);
}
